// WavLMAttention_46385646797440
// MI455X (gfx1250) — matrix operand order repaired and checked by reading, not compiled, not run on hardware
//
#include <hip/hip_runtime.h>
#include <hip/hip_bf16.h>

// ---------------------------------------------------------------------------
// WavLM attention for gfx1250 (MI455X): bf16 WMMA GEMMs + flash attention.
// B=4, S=1024, E=768, H=12, HD=64.  wave32, V_WMMA_F32_16X16X32_BF16,
// GLOBAL_LOAD_ASYNC_TO_LDS_B128 staging, DPP16 butterfly reductions.
// ---------------------------------------------------------------------------

#define B_ 4
#define S_ 1024
#define E_ 768
#define H_ 12
#define HD_ 64
#define N_ (B_ * S_)          // 4096 rows
#define NB_ 320               // relative-position buckets

typedef __attribute__((ext_vector_type(16))) __bf16 v16bf;
typedef __attribute__((ext_vector_type(8)))  float  v8f;

union ABFrag { unsigned int u[8]; v16bf v; };

__device__ __forceinline__ unsigned short f2bf(float x) {
  unsigned int u = __builtin_bit_cast(unsigned int, x);
  u += 0x7FFFu + ((u >> 16) & 1u);      // round-to-nearest-even
  return (unsigned short)(u >> 16);
}

// A-fragment (16x32, 16-bit) K index of the low half of VGPR r for this lane-half.
__device__ __forceinline__ int a_pair_k(int r, int half) {
  return 2 * r + 16 * half;
}

// ---- DPP16 xor-butterfly reductions over 16-lane row groups (wave32) ------
// xor1 = quad_perm[1,0,3,2] (0xB1), xor2 = quad_perm[2,3,0,1] (0x4E),
// xor7 = row_half_mirror (0x141), xor8 = row_ror:8 (0x128).
// Masks {1,2,7,8} span the 16-lane group with disjoint partials per stage.
template <int CTRL>
__device__ __forceinline__ float dpp_xor(float x) {
  int i = __builtin_bit_cast(int, x);
  i = __builtin_amdgcn_update_dpp(i, i, CTRL, 0xf, 0xf, true);
  return __builtin_bit_cast(float, i);
}
__device__ __forceinline__ float red16_max(float v) {
  v = fmaxf(v, dpp_xor<0xB1>(v));
  v = fmaxf(v, dpp_xor<0x4E>(v));
  v = fmaxf(v, dpp_xor<0x141>(v));
  v = fmaxf(v, dpp_xor<0x128>(v));
  return v;
}
__device__ __forceinline__ float red16_sum(float v) {
  v += dpp_xor<0xB1>(v);
  v += dpp_xor<0x4E>(v);
  v += dpp_xor<0x141>(v);
  v += dpp_xor<0x128>(v);
  return v;
}

// ---------------------------------------------------------------------------
// 1) position bias: out_pb[h][i][j] = rel_embed[bucket(j-i)][h]
// ---------------------------------------------------------------------------
__global__ void posbias_kernel(const float* __restrict__ rel_embed,
                               float* __restrict__ out_pb) {
  int idx = blockIdx.x * blockDim.x + threadIdx.x;   // over S*S
  if (idx >= S_ * S_) return;
  int i = idx >> 10, j = idx & (S_ - 1);
  int rel = j - i;
  const int nb = NB_ / 2;          // 160
  const int max_exact = nb / 2;    // 80
  int bucket = (rel > 0) ? nb : 0;
  int a = rel < 0 ? -rel : rel;
  int val;
  if (a < max_exact) {
    val = a;
  } else {
    float lg = __logf((float)a / (float)max_exact) * (1.0f / __logf(10.0f)) *
               (float)(nb - max_exact);
    int li = (int)((float)max_exact + lg);          // trunc toward zero (astype)
    val = li < (nb - 1) ? li : (nb - 1);
  }
  bucket += val;
#pragma unroll
  for (int h = 0; h < H_; ++h)
    out_pb[(size_t)h * S_ * S_ + idx] = rel_embed[bucket * H_ + h];
}

// ---------------------------------------------------------------------------
// 2) gate[b*H+h][s] = ga*(gb*gru_const[h]-1)+2
// ---------------------------------------------------------------------------
__global__ void gate_kernel(const float* __restrict__ hs,
                            const float* __restrict__ gw,   // (8, HD)
                            const float* __restrict__ gb,   // (8,)
                            const float* __restrict__ gc,   // (H,)
                            float* __restrict__ gate) {
  int idx = blockIdx.x * blockDim.x + threadIdx.x;   // (b, h, s)
  if (idx >= B_ * H_ * S_) return;
  int s = idx & (S_ - 1);
  int h = (idx >> 10) % H_;
  int b = idx / (H_ * S_);
  const float* x = hs + ((size_t)b * S_ + s) * E_ + h * HD_;
  float p[8];
#pragma unroll
  for (int j = 0; j < 8; ++j) {
    float acc = gb[j];
    const float* w = gw + j * HD_;
    for (int d = 0; d < HD_; ++d) acc += x[d] * w[d];
    p[j] = acc;
  }
  float p0 = p[0] + p[1] + p[2] + p[3];
  float p1 = p[4] + p[5] + p[6] + p[7];
  float ga = 1.0f / (1.0f + __expf(-p0));
  float gbv = 1.0f / (1.0f + __expf(-p1));
  gate[idx] = ga * (gbv * gc[h] - 1.0f) + 2.0f;
}

// ---------------------------------------------------------------------------
// 3) f32 -> bf16 convert
// ---------------------------------------------------------------------------
__global__ void cvt_kernel(const float* __restrict__ x,
                           unsigned short* __restrict__ y, int n) {
  int i = blockIdx.x * blockDim.x + threadIdx.x;
  if (i < n) y[i] = f2bf(x[i]);
}

// ---------------------------------------------------------------------------
// 4) GEMM: Y = X(bf16, N x 768) @ W(f32, 768 x 768)^T + bias
// MODE 0: q -> scale 0.125, store bf16 (B,H,S,HD)
// MODE 1: k -> store bf16 (B,H,S,HD)
// MODE 2: v -> store bf16 transposed (B,H,HD,S)
// MODE 3: out -> store f32 row-major (N, E)
// Block: 256 thr (8 waves), tile 128x128, BK=32.
// X tile staged with GLOBAL_LOAD_ASYNC_TO_LDS_B128 (ASYNCcnt path); the W
// tile must pass through VALU for the f32->bf16 convert, so it overlaps.
// ---------------------------------------------------------------------------
template <int MODE>
__global__ __launch_bounds__(256) void gemm_bf16(
    const unsigned short* __restrict__ X, const float* __restrict__ W,
    const float* __restrict__ bias, unsigned short* __restrict__ Yh,
    float* __restrict__ Yf) {
  constexpr int LDX = 40;                         // padded LDS row stride (halves)
  __shared__ unsigned short Xs[128 * LDX];
  __shared__ unsigned short Ws[128 * LDX];
  const int tid  = threadIdx.x;
  const int lane = tid & 31, wave = tid >> 5;
  const int row0 = blockIdx.x * 128;
  const int col0 = blockIdx.y * 128;
  const int wm = (wave & 3) * 32;                 // 32 M rows per wave
  const int wn = (wave >> 2) * 64;                // 64 N cols per wave
  const int mrow = lane & 15, half = lane >> 4;

  v8f acc[2][4];
#pragma unroll
  for (int i = 0; i < 2; ++i)
#pragma unroll
    for (int j = 0; j < 4; ++j)
#pragma unroll
      for (int e = 0; e < 8; ++e) acc[i][j][e] = 0.0f;

  for (int k0 = 0; k0 < E_; k0 += 32) {
    // stage X tile (128x32 bf16): async global -> LDS, 16B per lane issue
#pragma unroll
    for (int i = 0; i < 2; ++i) {
      int off = (tid * 2 + i) * 8;                // half offset in tile
      int r = off >> 5, c = off & 31;
      const unsigned short* gsrc = X + (size_t)(row0 + r) * E_ + k0 + c;
      unsigned lds_dst = (unsigned)(uintptr_t)(Xs + r * LDX + c);
      asm volatile("global_load_async_to_lds_b128 %0, %1, off"
                   :: "v"(lds_dst), "v"(gsrc)
                   : "memory");
    }
    // stage W tile (128 rows of output-cols x 32 k), f32 -> bf16
#pragma unroll
    for (int i = 0; i < 4; ++i) {
      int off = (tid * 4 + i) * 4;                // float offset in tile
      int r = off >> 5, c = off & 31;
      float4 d = *(const float4*)(W + (size_t)(col0 + r) * E_ + k0 + c);
      unsigned short* dst = Ws + r * LDX + c;
      dst[0] = f2bf(d.x); dst[1] = f2bf(d.y);
      dst[2] = f2bf(d.z); dst[3] = f2bf(d.w);
    }
    if (k0 + 32 < E_) {   // hint next tiles into cache (global_prefetch_b8)
      __builtin_prefetch(X + (size_t)(row0 + (tid >> 1)) * E_ + k0 + 32, 0, 0);
      __builtin_prefetch(W + (size_t)(col0 + (tid >> 1)) * E_ + k0 + 32, 0, 0);
    }
    asm volatile("s_wait_asynccnt 0x0" ::: "memory");
    __syncthreads();

    ABFrag afr[2], bfr[4];
#pragma unroll
    for (int mt = 0; mt < 2; ++mt)
#pragma unroll
      for (int r = 0; r < 8; ++r)
        afr[mt].u[r] = *(const unsigned int*)(
            Xs + (wm + mt * 16 + mrow) * LDX + a_pair_k(r, half));
#pragma unroll
    for (int nt = 0; nt < 4; ++nt)
#pragma unroll
      for (int r = 0; r < 8; ++r)
        bfr[nt].u[r] = *(const unsigned int*)(
            Ws + (wn + nt * 16 + mrow) * LDX + 2 * r + 16 * half);
#pragma unroll
    for (int mt = 0; mt < 2; ++mt)
#pragma unroll
      for (int nt = 0; nt < 4; ++nt)
        acc[mt][nt] = __builtin_amdgcn_wmma_f32_16x16x32_bf16(
            false, afr[mt].v, false, bfr[nt].v, (short)0, acc[mt][nt], false,
            false);
    __syncthreads();
  }

  // epilogue
#pragma unroll
  for (int mt = 0; mt < 2; ++mt)
#pragma unroll
    for (int nt = 0; nt < 4; ++nt)
#pragma unroll
      for (int r = 0; r < 8; ++r) {
        int grow = row0 + wm + mt * 16 + r + 8 * half;   // token index n
        int gcol = col0 + wn + nt * 16 + mrow;           // feature index e
        float y = acc[mt][nt][r] + bias[gcol];
        if (MODE == 3) {
          Yf[(size_t)grow * E_ + gcol] = y;
        } else {
          int b = grow >> 10, s = grow & (S_ - 1);
          int h = gcol >> 6, d = gcol & (HD_ - 1);
          if (MODE == 0) {
            y *= 0.125f;                                  // HD^-0.5
            Yh[((size_t)(b * H_ + h) * S_ + s) * HD_ + d] = f2bf(y);
          } else if (MODE == 1) {
            Yh[((size_t)(b * H_ + h) * S_ + s) * HD_ + d] = f2bf(y);
          } else {                                        // v transposed
            Yh[((size_t)(b * H_ + h) * HD_ + d) * S_ + s] = f2bf(y);
          }
        }
      }
}

// ---------------------------------------------------------------------------
// 5) flash attention per (b,h): softmax(QK^T + gate*pos_bias) @ V
// grid.x = S/128 q-tiles, grid.y = B*H; 8 waves, 16 q-rows per wave.
// ---------------------------------------------------------------------------
__global__ __launch_bounds__(256) void attn_kernel(
    const unsigned short* __restrict__ Q,    // (B*H, S, HD) bf16 pre-scaled
    const unsigned short* __restrict__ K,    // (B*H, S, HD) bf16
    const unsigned short* __restrict__ VT,   // (B*H, HD, S) bf16
    const float* __restrict__ gate,          // (B*H, S)
    const float* __restrict__ PB,            // (H, S, S) f32 (from d_out tail)
    unsigned short* __restrict__ CTX) {      // (B, S, H, HD) bf16
  const int bh = blockIdx.y;
  const int h = bh % H_;
  const int b = bh / H_;
  const int lane = threadIdx.x & 31, wave = threadIdx.x >> 5;
  const int q0 = blockIdx.x * 128 + wave * 16;
  const int mrow = lane & 15, half = lane >> 4;

  __shared__ float gate_s[128];
  __shared__ unsigned short Ps[8][16 * 36];            // per-wave P staging
  if (threadIdx.x < 128)
    gate_s[threadIdx.x] = gate[(size_t)bh * S_ + blockIdx.x * 128 + threadIdx.x];
  __syncthreads();

  // Q A-fragments for the two 32-wide K chunks of HD=64
  ABFrag qa[2];
  const unsigned short* qrow = Q + ((size_t)bh * S_ + q0 + mrow) * HD_;
#pragma unroll
  for (int c = 0; c < 2; ++c)
#pragma unroll
    for (int r = 0; r < 8; ++r)
      qa[c].u[r] =
          *(const unsigned int*)(qrow + c * 32 + a_pair_k(r, half));

  float rm[8], l[8];
  v8f O[4];
#pragma unroll
  for (int r = 0; r < 8; ++r) { rm[r] = -1e30f; l[r] = 0.0f; }
#pragma unroll
  for (int t = 0; t < 4; ++t)
#pragma unroll
    for (int e = 0; e < 8; ++e) O[t][e] = 0.0f;

  const float* pbbase = PB + (size_t)h * S_ * S_;
  unsigned short* psw = &Ps[wave][0];

  for (int j0 = 0; j0 < S_; j0 += 32) {
    // ---- scores for kv columns [j0, j0+32) : two 16-col tiles
    v8f sc[2];
#pragma unroll
    for (int t = 0; t < 2; ++t) {
#pragma unroll
      for (int e = 0; e < 8; ++e) sc[t][e] = 0.0f;
      const unsigned short* krow =
          K + ((size_t)bh * S_ + j0 + t * 16 + mrow) * HD_;
#pragma unroll
      for (int c = 0; c < 2; ++c) {
        ABFrag kb;
#pragma unroll
        for (int r = 0; r < 8; ++r)
          kb.u[r] = *(const unsigned int*)(krow + c * 32 + 2 * r + 16 * half);
        sc[t] = __builtin_amdgcn_wmma_f32_16x16x32_bf16(
            false, qa[c].v, false, kb.v, (short)0, sc[t], false, false);
      }
      // gated relative-position bias
#pragma unroll
      for (int r = 0; r < 8; ++r) {
        int m = r + 8 * half;
        float g = gate_s[wave * 16 + m];
        sc[t][r] +=
            g * pbbase[(size_t)(q0 + m) * S_ + j0 + t * 16 + mrow];
      }
    }
    // ---- online softmax (rows live in 16-lane halves, DPP butterflies)
    float nm[8], alpha[8];
#pragma unroll
    for (int r = 0; r < 8; ++r) {
      float v = red16_max(fmaxf(sc[0][r], sc[1][r]));
      nm[r] = fmaxf(rm[r], v);
      alpha[r] = __expf(rm[r] - nm[r]);
      rm[r] = nm[r];
    }
#pragma unroll
    for (int t = 0; t < 2; ++t)
#pragma unroll
      for (int r = 0; r < 8; ++r) sc[t][r] = __expf(sc[t][r] - nm[r]);
#pragma unroll
    for (int r = 0; r < 8; ++r) {
      float s = red16_sum(sc[0][r] + sc[1][r]);
      l[r] = l[r] * alpha[r] + s;
    }
#pragma unroll
    for (int t = 0; t < 4; ++t)
#pragma unroll
      for (int r = 0; r < 8; ++r) O[t][r] *= alpha[r];

    // ---- re-layout P (C-frag -> A-frag) through per-wave LDS
#pragma unroll
    for (int t = 0; t < 2; ++t)
#pragma unroll
      for (int r = 0; r < 8; ++r)
        psw[(r + 8 * half) * 36 + t * 16 + mrow] = f2bf(sc[t][r]);
    ABFrag pa;                       // LDS ops of one wave are in order
#pragma unroll
    for (int r = 0; r < 8; ++r)
      pa.u[r] = *(const unsigned int*)(psw + mrow * 36 + a_pair_k(r, half));

    // ---- context accumulate over 4 d-tiles of HD=64
#pragma unroll
    for (int t = 0; t < 4; ++t) {
      ABFrag vb;
      const unsigned short* vrow =
          VT + ((size_t)bh * HD_ + t * 16 + mrow) * S_ + j0;
#pragma unroll
      for (int r = 0; r < 8; ++r)
        vb.u[r] = *(const unsigned int*)(vrow + 2 * r + 16 * half);
      O[t] = __builtin_amdgcn_wmma_f32_16x16x32_bf16(
          false, pa.v, false, vb.v, (short)0, O[t], false, false);
    }
  }

  // ---- normalize and store ctx as bf16 row-major (B,S,H,HD)
#pragma unroll
  for (int r = 0; r < 8; ++r) {
    int m = r + 8 * half;
    float inv = 1.0f / l[r];
    int s = q0 + m;
#pragma unroll
    for (int t = 0; t < 4; ++t) {
      int d = t * 16 + mrow;
      CTX[((size_t)b * S_ + s) * E_ + h * HD_ + d] = f2bf(O[t][r] * inv);
    }
  }
}

// ---------------------------------------------------------------------------
// launcher
// ---------------------------------------------------------------------------
extern "C" void kernel_launch(void* const* d_in, const int* in_sizes, int n_in,
                              void* d_out, int out_size, void* d_ws,
                              size_t ws_size, hipStream_t stream) {
  const float* hidden = (const float*)d_in[0];
  const float* q_w = (const float*)d_in[1];
  const float* q_b = (const float*)d_in[2];
  const float* k_w = (const float*)d_in[3];
  const float* k_b = (const float*)d_in[4];
  const float* v_w = (const float*)d_in[5];
  const float* v_b = (const float*)d_in[6];
  const float* out_w = (const float*)d_in[7];
  const float* out_b = (const float*)d_in[8];
  const float* gru_const = (const float*)d_in[9];
  const float* gru_w = (const float*)d_in[10];
  const float* gru_b = (const float*)d_in[11];
  const float* rel_embed = (const float*)d_in[12];

  float* out_attn = (float*)d_out;                 // (B,S,E)
  float* out_pb = out_attn + (size_t)N_ * E_;      // (H,S,S)

  // workspace carving (bytes)
  char* ws = (char*)d_ws;
  const size_t NE2 = (size_t)N_ * E_ * sizeof(unsigned short);
  unsigned short* x_bf = (unsigned short*)(ws);
  unsigned short* q_bf = (unsigned short*)(ws + NE2);
  unsigned short* k_bf = (unsigned short*)(ws + 2 * NE2);
  unsigned short* vT_bf = (unsigned short*)(ws + 3 * NE2);
  unsigned short* ctx_bf = (unsigned short*)(ws + 4 * NE2);
  float* gate_ws = (float*)(ws + 5 * NE2);         // B*H*S floats

  // 1) position bias (also the second output)
  posbias_kernel<<<(S_ * S_ + 255) / 256, 256, 0, stream>>>(rel_embed, out_pb);
  // 2) gates
  gate_kernel<<<(B_ * H_ * S_ + 255) / 256, 256, 0, stream>>>(
      hidden, gru_w, gru_b, gru_const, gate_ws);
  // 3) hidden -> bf16
  cvt_kernel<<<((int)((size_t)N_ * E_) + 255) / 256, 256, 0, stream>>>(
      hidden, x_bf, N_ * E_);
  // 4) projections
  dim3 gg(N_ / 128, E_ / 128);
  gemm_bf16<0><<<gg, 256, 0, stream>>>(x_bf, q_w, q_b, q_bf, nullptr);
  gemm_bf16<1><<<gg, 256, 0, stream>>>(x_bf, k_w, k_b, k_bf, nullptr);
  gemm_bf16<2><<<gg, 256, 0, stream>>>(x_bf, v_w, v_b, vT_bf, nullptr);
  // 5) attention
  dim3 ga(S_ / 128, B_ * H_);
  attn_kernel<<<ga, 256, 0, stream>>>(q_bf, k_bf, vT_bf, gate_ws, out_pb,
                                      ctx_bf);
  // 6) output projection -> attn_output
  gemm_bf16<3><<<gg, 256, 0, stream>>>(ctx_bf, out_w, out_b, nullptr, out_attn);
}
